// LoRANode2Edge2NodeBlockDGLSymmetry_83734682402872
// MI455X (gfx1250) — compile-verified, matrix operand order checked by reading, not run on hardware
//
#include <hip/hip_runtime.h>
#include <hip/hip_bf16.h>

typedef __attribute__((ext_vector_type(16))) _Float16 v16h;
typedef __attribute__((ext_vector_type(8)))  float    v8f;

#define D 128
#define LN_EPS 1e-5f

// ---------------------------------------------------------------------------
// Utility kernels
// ---------------------------------------------------------------------------
__global__ void convert_weights_kernel(const float* __restrict__ a,
                                       const float* __restrict__ b,
                                       const float* __restrict__ c,
                                       const float* __restrict__ d,
                                       _Float16* __restrict__ out) {
    int i = blockIdx.x * blockDim.x + threadIdx.x;
    if (i >= 4 * D * D) return;
    const float* srcs[4] = {a, b, c, d};
    out[i] = (_Float16)srcs[i >> 14][i & (D * D - 1)];
}

__global__ void zero_f32_kernel(float* __restrict__ p, long n) {
    long i = blockIdx.x * (long)blockDim.x + threadIdx.x;
    if (i < n) p[i] = 0.0f;
}

// ---------------------------------------------------------------------------
// Phase 1: per-edge  add_feat = LN(SiLU((src+tgt)@Ws2e^T + edge@We2e^T))
//          edge_out = edge_feat + add_feat
//          sums[tgt] += add_feat ; counts[tgt] += 1   (scatter for mean)
// One wave32 handles 16 edges x 128 dims via 8 x (16x16) WMMA tiles.
// ---------------------------------------------------------------------------
__global__ __launch_bounds__(256)
void edge_phase_kernel(const float* __restrict__ src_feat,
                       const float* __restrict__ tgt_feat,
                       const float* __restrict__ edge_feat,
                       const _Float16* __restrict__ W0,   // Ws2e  [n][k] f16
                       const _Float16* __restrict__ W1,   // We2e  [n][k] f16
                       const float* __restrict__ gamma,
                       const float* __restrict__ beta,
                       const int* __restrict__ src_idx,
                       const int* __restrict__ tgt_idx,
                       float* __restrict__ edge_out,
                       float* __restrict__ sums,
                       float* __restrict__ counts,
                       int n_edge) {
    const int lane = threadIdx.x & 31;
    const int wave = threadIdx.x >> 5;
    const int tile = blockIdx.x * 8 + wave;
    const int e0   = tile * 16;
    if (e0 >= n_edge) return;

    const int g   = lane >> 4;    // K-half selector for A/B layout
    const int i16 = lane & 15;    // this lane's row (A) / column (B,C,D)

    // Gather endpoints for this lane's row.
    const int si = src_idx[e0 + i16];
    const int ti = tgt_idx[e0 + i16];
    const float* __restrict__ srow = src_feat  + (size_t)si * D;
    const float* __restrict__ trow = tgt_feat  + (size_t)ti * D;
    const float* __restrict__ erow = edge_feat + (size_t)(e0 + i16) * D;

    // A operands for all 4 K-steps: documented 16-bit A layout
    //   elems 0..7  -> K = kb*32 + 8*g + j
    //   elems 8..15 -> K = kb*32 + 16 + 8*g + j
    v16h A1[4], A2[4];
#pragma unroll
    for (int kb = 0; kb < 4; ++kb) {
        const int k0 = kb * 32 + 8 * g;
#pragma unroll
        for (int j = 0; j < 8; ++j) {
            A1[kb][j]     = (_Float16)(srow[k0 + j]      + trow[k0 + j]);
            A1[kb][8 + j] = (_Float16)(srow[k0 + 16 + j] + trow[k0 + 16 + j]);
            A2[kb][j]     = (_Float16)erow[k0 + j];
            A2[kb][8 + j] = (_Float16)erow[k0 + 16 + j];
        }
    }

    // 8 output tiles of 16x16, each = sum over 4 K-steps of two GEMMs.
    v8f acc[8];
#pragma unroll
    for (int nt = 0; nt < 8; ++nt) {
        v8f c = {};
        const _Float16* __restrict__ w0r = W0 + (size_t)(nt * 16 + i16) * D;
        const _Float16* __restrict__ w1r = W1 + (size_t)(nt * 16 + i16) * D;
#pragma unroll
        for (int kb = 0; kb < 4; ++kb) {
            const int k0 = kb * 32 + 8 * g;
            v16h b0, b1;
#pragma unroll
            for (int j = 0; j < 8; ++j) {
                b0[j] = w0r[k0 + j];  b0[8 + j] = w0r[k0 + 16 + j];
                b1[j] = w1r[k0 + j];  b1[8 + j] = w1r[k0 + 16 + j];
            }
            c = __builtin_amdgcn_wmma_f32_16x16x32_f16(false, A1[kb], false, b0,
                                                       (short)0, c, false, false);
            c = __builtin_amdgcn_wmma_f32_16x16x32_f16(false, A2[kb], false, b1,
                                                       (short)0, c, false, false);
        }
        acc[nt] = c;
    }

    // SiLU in-register, accumulate row statistics.
    // C/D layout: VGPR r, lane group g -> row m = r + 8*g ; column = nt*16 + i16.
    float rowsum[8], rowsq[8];
#pragma unroll
    for (int r = 0; r < 8; ++r) { rowsum[r] = 0.0f; rowsq[r] = 0.0f; }
#pragma unroll
    for (int nt = 0; nt < 8; ++nt) {
#pragma unroll
        for (int r = 0; r < 8; ++r) {
            float y  = acc[nt][r];
            float sv = y / (1.0f + __expf(-y));   // SiLU
            acc[nt][r] = sv;
            rowsum[r] += sv;
            rowsq[r]  += sv * sv;
        }
    }
    // Cross-lane reduce within each 16-lane half (rows differ between halves).
#pragma unroll
    for (int m = 1; m < 16; m <<= 1) {
#pragma unroll
        for (int r = 0; r < 8; ++r) {
            rowsum[r] += __shfl_xor(rowsum[r], m, 32);
            rowsq[r]  += __shfl_xor(rowsq[r],  m, 32);
        }
    }
    float mean[8], rinv[8];
#pragma unroll
    for (int r = 0; r < 8; ++r) {
        mean[r]   = rowsum[r] * (1.0f / D);
        float var = rowsq[r] * (1.0f / D) - mean[r] * mean[r];
        rinv[r]   = rsqrtf(var + LN_EPS);
    }

    float gam[8], bet[8];
#pragma unroll
    for (int nt = 0; nt < 8; ++nt) {
        gam[nt] = gamma[nt * 16 + i16];
        bet[nt] = beta [nt * 16 + i16];
    }

    // Residual store + scatter-add of add_feat.
#pragma unroll
    for (int r = 0; r < 8; ++r) {
        const int m  = r + 8 * g;
        const int e  = e0 + m;
        const int tg = __shfl(ti, m, 32);   // lane m (<16) holds row m's tgt idx
        const float* __restrict__ er = edge_feat + (size_t)e * D;
        float* __restrict__       orow = edge_out + (size_t)e * D;
        float* __restrict__       srow2 = sums + (size_t)tg * D;
#pragma unroll
        for (int nt = 0; nt < 8; ++nt) {
            const int n = nt * 16 + i16;
            float add = (acc[nt][r] - mean[r]) * rinv[r] * gam[nt] + bet[nt];
            orow[n] = er[n] + add;
            atomicAdd(&srow2[n], add);
        }
    }
    if (g == 0) atomicAdd(&counts[ti], 1.0f);
}

// ---------------------------------------------------------------------------
// Phase 2: per-target-node  agg = sums/max(counts,1)
//          add_t = LN(SiLU(agg@We2t^T + tgt@Wt2t^T)) ; tgt_out = tgt + add_t
// ---------------------------------------------------------------------------
__global__ __launch_bounds__(256)
void node_phase_kernel(const float* __restrict__ sums,
                       const float* __restrict__ counts,
                       const float* __restrict__ tgt_feat,
                       const _Float16* __restrict__ W2,   // We2t
                       const _Float16* __restrict__ W3,   // Wt2t
                       const float* __restrict__ gamma,
                       const float* __restrict__ beta,
                       float* __restrict__ tgt_out,
                       int n_tgt) {
    const int lane = threadIdx.x & 31;
    const int wave = threadIdx.x >> 5;
    const int tile = blockIdx.x * 8 + wave;
    const int v0   = tile * 16;
    if (v0 >= n_tgt) return;

    const int g   = lane >> 4;
    const int i16 = lane & 15;

    const float cnt = counts[v0 + i16];
    const float inv = 1.0f / fmaxf(cnt, 1.0f);
    const float* __restrict__ arow = sums     + (size_t)(v0 + i16) * D;
    const float* __restrict__ trow = tgt_feat + (size_t)(v0 + i16) * D;

    v16h A1[4], A2[4];
#pragma unroll
    for (int kb = 0; kb < 4; ++kb) {
        const int k0 = kb * 32 + 8 * g;
#pragma unroll
        for (int j = 0; j < 8; ++j) {
            A1[kb][j]     = (_Float16)(arow[k0 + j]      * inv);
            A1[kb][8 + j] = (_Float16)(arow[k0 + 16 + j] * inv);
            A2[kb][j]     = (_Float16)trow[k0 + j];
            A2[kb][8 + j] = (_Float16)trow[k0 + 16 + j];
        }
    }

    v8f acc[8];
#pragma unroll
    for (int nt = 0; nt < 8; ++nt) {
        v8f c = {};
        const _Float16* __restrict__ w2r = W2 + (size_t)(nt * 16 + i16) * D;
        const _Float16* __restrict__ w3r = W3 + (size_t)(nt * 16 + i16) * D;
#pragma unroll
        for (int kb = 0; kb < 4; ++kb) {
            const int k0 = kb * 32 + 8 * g;
            v16h b0, b1;
#pragma unroll
            for (int j = 0; j < 8; ++j) {
                b0[j] = w2r[k0 + j];  b0[8 + j] = w2r[k0 + 16 + j];
                b1[j] = w3r[k0 + j];  b1[8 + j] = w3r[k0 + 16 + j];
            }
            c = __builtin_amdgcn_wmma_f32_16x16x32_f16(false, A1[kb], false, b0,
                                                       (short)0, c, false, false);
            c = __builtin_amdgcn_wmma_f32_16x16x32_f16(false, A2[kb], false, b1,
                                                       (short)0, c, false, false);
        }
        acc[nt] = c;
    }

    float rowsum[8], rowsq[8];
#pragma unroll
    for (int r = 0; r < 8; ++r) { rowsum[r] = 0.0f; rowsq[r] = 0.0f; }
#pragma unroll
    for (int nt = 0; nt < 8; ++nt) {
#pragma unroll
        for (int r = 0; r < 8; ++r) {
            float y  = acc[nt][r];
            float sv = y / (1.0f + __expf(-y));
            acc[nt][r] = sv;
            rowsum[r] += sv;
            rowsq[r]  += sv * sv;
        }
    }
#pragma unroll
    for (int m = 1; m < 16; m <<= 1) {
#pragma unroll
        for (int r = 0; r < 8; ++r) {
            rowsum[r] += __shfl_xor(rowsum[r], m, 32);
            rowsq[r]  += __shfl_xor(rowsq[r],  m, 32);
        }
    }
    float mean[8], rinv[8];
#pragma unroll
    for (int r = 0; r < 8; ++r) {
        mean[r]   = rowsum[r] * (1.0f / D);
        float var = rowsq[r] * (1.0f / D) - mean[r] * mean[r];
        rinv[r]   = rsqrtf(var + LN_EPS);
    }

    float gam[8], bet[8];
#pragma unroll
    for (int nt = 0; nt < 8; ++nt) {
        gam[nt] = gamma[nt * 16 + i16];
        bet[nt] = beta [nt * 16 + i16];
    }

#pragma unroll
    for (int r = 0; r < 8; ++r) {
        const int m    = r + 8 * g;
        const int node = v0 + m;
        const float* __restrict__ tr = tgt_feat + (size_t)node * D;
        float* __restrict__       orow = tgt_out + (size_t)node * D;
#pragma unroll
        for (int nt = 0; nt < 8; ++nt) {
            const int n = nt * 16 + i16;
            float add = (acc[nt][r] - mean[r]) * rinv[r] * gam[nt] + bet[nt];
            orow[n] = tr[n] + add;
        }
    }
}

// ---------------------------------------------------------------------------
// Launch
// ---------------------------------------------------------------------------
extern "C" void kernel_launch(void* const* d_in, const int* in_sizes, int n_in,
                              void* d_out, int out_size, void* d_ws, size_t ws_size,
                              hipStream_t stream) {
    const float* src_feat  = (const float*)d_in[0];
    const float* tgt_feat  = (const float*)d_in[1];
    const float* edge_feat = (const float*)d_in[2];
    const float* Ws2e      = (const float*)d_in[3];
    const float* We2e      = (const float*)d_in[4];
    const float* We2t      = (const float*)d_in[5];
    const float* Wt2t      = (const float*)d_in[6];
    const float* gamma1    = (const float*)d_in[7];
    const float* beta1     = (const float*)d_in[8];
    const float* gamma2    = (const float*)d_in[9];
    const float* beta2     = (const float*)d_in[10];
    const int*   src_idx   = (const int*)d_in[11];
    const int*   tgt_idx   = (const int*)d_in[12];

    const int E    = in_sizes[11];        // 640000
    const int NTGT = in_sizes[1] / D;     // 40000

    // Workspace layout: [4 x 128 x 128 f16 weights][sums f32 NTGT*D][counts f32 NTGT]
    _Float16* Wh     = (_Float16*)d_ws;
    float*    sums   = (float*)((char*)d_ws + (size_t)4 * D * D * sizeof(_Float16));
    float*    counts = sums + (size_t)NTGT * D;

    float* edge_out = (float*)d_out;
    float* tgt_out  = edge_out + (size_t)E * D;

    // 1) Convert the four weight matrices to f16 (row-major [out][in]).
    convert_weights_kernel<<<(4 * D * D + 255) / 256, 256, 0, stream>>>(
        Ws2e, We2e, We2t, Wt2t, Wh);

    // 2) Zero the scatter accumulators (ws is poisoned; must re-zero every call).
    const long nzero = (long)NTGT * D + NTGT;
    zero_f32_kernel<<<(int)((nzero + 255) / 256), 256, 0, stream>>>(sums, nzero);

    // 3) Edge phase: WMMA GEMMs + SiLU + LN + residual + atomic scatter.
    const int etiles = (E + 15) / 16;
    edge_phase_kernel<<<(etiles + 7) / 8, 256, 0, stream>>>(
        src_feat, tgt_feat, edge_feat, Wh, Wh + (size_t)D * D,
        gamma1, beta1, src_idx, tgt_idx, edge_out, sums, counts, E);

    // 4) Node phase: mean, WMMA GEMMs + SiLU + LN + residual.
    const int vtiles = (NTGT + 15) / 16;
    node_phase_kernel<<<(vtiles + 7) / 8, 256, 0, stream>>>(
        sums, counts, tgt_feat, Wh + (size_t)2 * D * D, Wh + (size_t)3 * D * D,
        gamma2, beta2, tgt_out, NTGT);
}
